// GRU_1606317769112
// MI455X (gfx1250) — matrix operand order repaired and checked by reading, not compiled, not run on hardware
//
#include <hip/hip_runtime.h>
#include <cstddef>

// GRU forward for MI455X (gfx1250): bf16 WMMA (v_wmma_f32_16x16x32_bf16),
// L2-resident recurrent weights, ping-pong bf16 hidden state.
// Round 2: split K-loop into input-phase / hidden-phase straight-line loops
// (no per-iteration cndmask/branch on the critical sequential path).

#define BDIM 64
#define SDIM 512
#define IDIM 512
#define HDIM 1024
#define KC   (IDIM + HDIM)   // 1536 concatenated K

typedef __attribute__((ext_vector_type(16))) __bf16        v16bf;
typedef __attribute__((ext_vector_type(8)))  float         v8f;
typedef __attribute__((ext_vector_type(8)))  unsigned int  v8u;

__device__ __forceinline__ unsigned short f2bf(float f) {
    unsigned int u = __float_as_uint(f);
    u += 0x7FFFu + ((u >> 16) & 1u);          // round-to-nearest-even
    return (unsigned short)(u >> 16);
}

// ---- A fragment: 16x32 bf16, row M = lane&15.
// lane<16 : elems 0..7 -> K=kb+0..7,  elems 8..15 -> K=kb+16..23
// lane>=16: elems 0..7 -> K=kb+8..15, elems 8..15 -> K=kb+24..31
__device__ __forceinline__ v16bf load_frag_a(const unsigned short* __restrict__ row,
                                             int kb, int lane) {
    int c0 = kb + ((lane & 16) ? 8 : 0);
    uint4 lo = *reinterpret_cast<const uint4*>(row + c0);
    uint4 hi = *reinterpret_cast<const uint4*>(row + c0 + 16);
    v8u r;
    r[0] = lo.x; r[1] = lo.y; r[2] = lo.z; r[3] = lo.w;
    r[4] = hi.x; r[5] = hi.y; r[6] = hi.z; r[7] = hi.w;
    return __builtin_bit_cast(v16bf, r);
}

// ---- B fragment: 32x16 bf16, column N = lane&15 (a row of W, contiguous K).
// lane<16 : K = kb+0..7, kb+16..23 ; lane>=16 : K = kb+8..15, kb+24..31
__device__ __forceinline__ v16bf load_frag_b(const unsigned short* __restrict__ row,
                                             int kb, int lane) {
    int c0 = kb + ((lane & 16) ? 8 : 0);
    const uint4* p = reinterpret_cast<const uint4*>(row + c0);
    uint4 lo = p[0];
    uint4 hi = p[2];
    v8u r;
    r[0] = lo.x; r[1] = lo.y; r[2] = lo.z; r[3] = lo.w;
    r[4] = hi.x; r[5] = hi.y; r[6] = hi.z; r[7] = hi.w;
    return __builtin_bit_cast(v16bf, r);
}

__device__ __forceinline__ v8f wmma_bf16(v16bf a, v16bf b, v8f c) {
    return __builtin_amdgcn_wmma_f32_16x16x32_bf16(
        false, a, false, b, (short)0, c, false, false);
}

// -------------------- one-time prep kernels --------------------

// Pack [w_i (H x I) | w_h (H x H)] -> bf16 (H x KC), contiguous K per row.
__global__ void pack_w_kernel(const float* __restrict__ wi,
                              const float* __restrict__ wh,
                              unsigned short* __restrict__ wc) {
    int idx = blockIdx.x * blockDim.x + threadIdx.x;
    const int n = HDIM * KC;
    if (idx >= n) return;
    int h = idx / KC;
    int k = idx - h * KC;
    float v = (k < IDIM) ? wi[(size_t)h * IDIM + k]
                         : wh[(size_t)h * HDIM + (k - IDIM)];
    wc[idx] = f2bf(v);
}

__global__ void pack_x_kernel(const float* __restrict__ x,
                              unsigned short* __restrict__ xb, int n) {
    int idx = blockIdx.x * blockDim.x + threadIdx.x;
    if (idx < n) xb[idx] = f2bf(x[idx]);
}

// out[:,0,:] = h0 ; hb0 = bf16(h0)
__global__ void init_h_kernel(const float* __restrict__ h0,
                              float* __restrict__ out,
                              unsigned short* __restrict__ hb0) {
    int idx = blockIdx.x * blockDim.x + threadIdx.x;
    const int n = BDIM * HDIM;
    if (idx >= n) return;
    int b = idx / HDIM;
    int h = idx - b * HDIM;
    float v = h0[idx];
    out[((size_t)b * SDIM) * HDIM + h] = v;
    hb0[idx] = f2bf(v);
}

// h_last = out[:, S-1, :] appended after outputs
__global__ void tail_kernel(float* __restrict__ out) {
    int idx = blockIdx.x * blockDim.x + threadIdx.x;
    const int n = BDIM * HDIM;
    if (idx >= n) return;
    int b = idx / HDIM;
    int h = idx - b * HDIM;
    out[(size_t)BDIM * SDIM * HDIM + idx] =
        out[((size_t)b * SDIM + (SDIM - 1)) * HDIM + h];
}

// -------------------- recurrent step --------------------
// grid.x = HDIM/16 column tiles, 128 threads = 4 waves; wave w owns batch
// rows [16w, 16w+16). Four f32 accumulators per 16x16 tile:
//   sr = x*Wir^T + h*Whr^T ; sz = x*Wiz^T + h*Whz^T
//   sxn = x*Win^T          ; shn = h*Whn^T     (b_hn lives inside r*(...))
__global__ __launch_bounds__(128)
void gru_step_kernel(const unsigned short* __restrict__ xbf,
                     const unsigned short* __restrict__ wr,
                     const unsigned short* __restrict__ wz,
                     const unsigned short* __restrict__ wn,
                     const float* __restrict__ b_ir, const float* __restrict__ b_iz,
                     const float* __restrict__ b_in, const float* __restrict__ b_hr,
                     const float* __restrict__ b_hz, const float* __restrict__ b_hn,
                     const unsigned short* __restrict__ hprev, // (B,H) bf16
                     unsigned short* __restrict__ hnext,       // (B,H) bf16
                     float* __restrict__ out,                  // (B,S,H) f32
                     int t) {
    const int lane = threadIdx.x & 31;
    const int wave = threadIdx.x >> 5;
    const int m0 = wave << 4;           // batch tile base
    const int n0 = blockIdx.x << 4;     // hidden-column tile base
    const int col = lane & 15;

    const unsigned short* xrow = xbf + ((size_t)(m0 + col) * SDIM + t) * IDIM;
    const unsigned short* hrow = hprev + (size_t)(m0 + col) * HDIM;
    const unsigned short* wrrow = wr + (size_t)(n0 + col) * KC;
    const unsigned short* wzrow = wz + (size_t)(n0 + col) * KC;
    const unsigned short* wnrow = wn + (size_t)(n0 + col) * KC;
    // hidden-phase weight row bases (K offset IDIM.. in concatenated layout)
    const unsigned short* wrrowh = wrrow + IDIM;
    const unsigned short* wzrowh = wzrow + IDIM;
    const unsigned short* wnrowh = wnrow + IDIM;

    v8f sr = {}, sz = {}, sxn = {}, shn = {};

    // ---- phase 1: input projection, K = 0..IDIM ----
    for (int kb = 0; kb < IDIM; kb += 32) {
        __builtin_prefetch(wrrow + kb + 32, 0, 1);
        __builtin_prefetch(wzrow + kb + 32, 0, 1);
        __builtin_prefetch(wnrow + kb + 32, 0, 1);
        v16bf a   = load_frag_a(xrow, kb, lane);
        v16bf fbr = load_frag_b(wrrow, kb, lane);
        v16bf fbz = load_frag_b(wzrow, kb, lane);
        v16bf fbn = load_frag_b(wnrow, kb, lane);
        sr  = wmma_bf16(a, fbr, sr);
        sz  = wmma_bf16(a, fbz, sz);
        sxn = wmma_bf16(a, fbn, sxn);
    }

    // ---- phase 2: hidden projection, K = 0..HDIM (weights at +IDIM) ----
    for (int kb = 0; kb < HDIM; kb += 32) {
        if (kb + 32 < HDIM) {
            __builtin_prefetch(wrrowh + kb + 32, 0, 1);
            __builtin_prefetch(wzrowh + kb + 32, 0, 1);
            __builtin_prefetch(wnrowh + kb + 32, 0, 1);
        }
        v16bf a   = load_frag_a(hrow, kb, lane);
        v16bf fbr = load_frag_b(wrrowh, kb, lane);
        v16bf fbz = load_frag_b(wzrowh, kb, lane);
        v16bf fbn = load_frag_b(wnrowh, kb, lane);
        sr  = wmma_bf16(a, fbr, sr);
        sz  = wmma_bf16(a, fbz, sz);
        shn = wmma_bf16(a, fbn, shn);
    }

    // epilogue: C/D layout -> m = v + 8*(lane>=16), n = n0 + (lane&15)
    const int n = n0 + col;
    const int mhalf = (lane >> 4) << 3;
    const float bir = b_ir[n] + b_hr[n];
    const float biz = b_iz[n] + b_hz[n];
    const float bin = b_in[n];
    const float bhn = b_hn[n];

#pragma unroll
    for (int v = 0; v < 8; ++v) {
        const int b = m0 + mhalf + v;
        const float hp = out[((size_t)b * SDIM + (t - 1)) * HDIM + n];
        const float r  = 1.0f / (1.0f + __expf(-(sr[v] + bir)));
        const float z  = 1.0f / (1.0f + __expf(-(sz[v] + biz)));
        const float nn = tanhf(sxn[v] + bin + r * (shn[v] + bhn));
        const float hnew = (1.0f - z) * nn + z * hp;
        out[((size_t)b * SDIM + t) * HDIM + n] = hnew;
        hnext[(size_t)b * HDIM + n] = f2bf(hnew);
    }
}

// -------------------- launcher --------------------
extern "C" void kernel_launch(void* const* d_in, const int* in_sizes, int n_in,
                              void* d_out, int out_size, void* d_ws, size_t ws_size,
                              hipStream_t stream) {
    const float* inputs = (const float*)d_in[0];
    const float* h0     = (const float*)d_in[1];
    const float* w_ir   = (const float*)d_in[2];
    const float* w_iz   = (const float*)d_in[3];
    const float* w_in   = (const float*)d_in[4];
    const float* b_ir   = (const float*)d_in[5];
    const float* b_iz   = (const float*)d_in[6];
    const float* b_in   = (const float*)d_in[7];
    const float* w_hr   = (const float*)d_in[8];
    const float* w_hz   = (const float*)d_in[9];
    const float* w_hn   = (const float*)d_in[10];
    const float* b_hr   = (const float*)d_in[11];
    const float* b_hz   = (const float*)d_in[12];
    const float* b_hn   = (const float*)d_in[13];
    float* out = (float*)d_out;

    unsigned short* wr  = (unsigned short*)d_ws;
    unsigned short* wz  = wr + (size_t)HDIM * KC;
    unsigned short* wn  = wz + (size_t)HDIM * KC;
    unsigned short* xbf = wn + (size_t)HDIM * KC;
    unsigned short* hb0 = xbf + (size_t)BDIM * SDIM * IDIM;
    unsigned short* hb1 = hb0 + (size_t)BDIM * HDIM;

    {   // pack weights -> bf16 [Wi | Wh] per gate
        const int n = HDIM * KC;
        const int blk = 256, g = (n + blk - 1) / blk;
        pack_w_kernel<<<g, blk, 0, stream>>>(w_ir, w_hr, wr);
        pack_w_kernel<<<g, blk, 0, stream>>>(w_iz, w_hz, wz);
        pack_w_kernel<<<g, blk, 0, stream>>>(w_in, w_hn, wn);
    }
    {   // pack x -> bf16
        const int n = BDIM * SDIM * IDIM;
        pack_x_kernel<<<(n + 255) / 256, 256, 0, stream>>>(inputs, xbf, n);
    }
    {   // h0 -> out[:,0,:] and bf16 ping buffer
        const int n = BDIM * HDIM;
        init_h_kernel<<<(n + 255) / 256, 256, 0, stream>>>(h0, out, hb0);
    }

    for (int t = 1; t < SDIM; ++t) {
        const unsigned short* hp = ((t - 1) & 1) ? hb1 : hb0;
        unsigned short*       hn = ((t - 1) & 1) ? hb0 : hb1;
        gru_step_kernel<<<dim3(HDIM / 16), 128, 0, stream>>>(
            xbf, wr, wz, wn,
            b_ir, b_iz, b_in, b_hr, b_hz, b_hn,
            hp, hn, out, t);
    }

    {   // h_last
        const int n = BDIM * HDIM;
        tail_kernel<<<(n + 255) / 256, 256, 0, stream>>>(out);
    }
}
